// Block_34565896798239
// MI455X (gfx1250) — matrix operand order repaired and checked by reading, not compiled, not run on hardware
//
#include <hip/hip_runtime.h>
#include <hip/hip_bf16.h>

// ---------------------------------------------------------------------------
// Transformer block for MI455X (gfx1250, wave32, WMMA).
// fp32 reference; internals run bf16 WMMA (16x16x32) with fp32 accumulation.
// GEMM tiles staged with the Tensor Data Mover (tensor_load_to_lds, TENSORcnt,
// double-buffered LDS); attention K-blocks staged with async global->LDS
// (ASYNCcnt).
// ---------------------------------------------------------------------------

typedef __attribute__((ext_vector_type(16))) __bf16    v16bf;
typedef __attribute__((ext_vector_type(8)))  __bf16    v8bf;
typedef __attribute__((ext_vector_type(8)))  float     v8f;
typedef __attribute__((ext_vector_type(4)))  unsigned  u32x4;
typedef __attribute__((ext_vector_type(8)))  unsigned  u32x8;

__device__ __forceinline__ v16bf cat8(v8bf lo, v8bf hi) {
  return __builtin_shufflevector(lo, hi, 0,1,2,3,4,5,6,7,8,9,10,11,12,13,14,15);
}

__device__ __forceinline__ v8f wmma_bf16(v16bf a, v16bf b, v8f c) {
  // D = A(16x32 bf16) x B(32x16 bf16) + C(16x16 f32)
  return __builtin_amdgcn_wmma_f32_16x16x32_bf16(
      /*neg_a=*/false, a, /*neg_b=*/false, b,
      /*c_mod=*/(short)0, c, /*reuse_a=*/false, /*reuse_b=*/false);
}

// ---- async global->LDS (per-lane 16B, tracked by ASYNCcnt) ----------------
__device__ __forceinline__ void async_b128(unsigned lds_off, const void* gptr) {
  unsigned long long ga = (unsigned long long)gptr;
  asm volatile("global_load_async_to_lds_b128 %0, %1, off"
               :: "v"(lds_off), "v"(ga) : "memory");
}
__device__ __forceinline__ void wait_async_0() {
  asm volatile("s_wait_asynccnt 0x0" ::: "memory");
}

// ---- Tensor Data Mover: 2D tile global -> LDS (tracked by TENSORcnt) ------
// D# group0: count=1 | lds_addr | global_addr(57b) | type=2
// D# group1: data_size=2B, tensor_dim0=tile0, tensor_dim1=tile1,
//            tile_dim0=tile0, tile_dim1=tile1, tensor_dim0_stride=stride.
__device__ __forceinline__ void tdm_load_2d(unsigned lds_off, const void* gptr,
                                            unsigned tile0, unsigned tile1,
                                            unsigned stride_elems) {
  unsigned long long ga = (unsigned long long)gptr;
  u32x4 g0;
  g0[0] = 1u;                                              // count=1 (valid D#)
  g0[1] = lds_off;                                         // LDS byte address
  g0[2] = (unsigned)ga;                                    // global_addr[31:0]
  g0[3] = (unsigned)((ga >> 32) & 0x1FFFFFFu) | (2u << 30);// addr[56:32]|type=2
  u32x8 g1;
  g1[0] = 0x10000u;                    // wg_mask=0, data_size=1 (2 bytes)
  g1[1] = (tile0 & 0xFFFFu) << 16;     // tensor_dim0[15:0]
  g1[2] = (tile1 & 0xFFFFu) << 16;     // tensor_dim0[31:16]=0 | tensor_dim1[15:0]
  g1[3] = (tile0 << 16);               // tensor_dim1[31:16]=0 | tile_dim0
  g1[4] = tile1;                       // tile_dim1 | tile_dim2=0
  g1[5] = stride_elems;                // tensor_dim0_stride[31:0]
  g1[6] = 0u;                          // stride hi | dim1_stride lo
  g1[7] = 0u;
  asm volatile("tensor_load_to_lds %0, %1" :: "s"(g0), "s"(g1) : "memory");
}
__device__ __forceinline__ void wait_tensor_0() {
#if __has_builtin(__builtin_amdgcn_s_wait_tensorcnt)
  __builtin_amdgcn_s_wait_tensorcnt((short)0);
#else
  asm volatile("s_wait_tensorcnt 0x0" ::: "memory");
#endif
}
__device__ __forceinline__ void wait_tensor_2() {
#if __has_builtin(__builtin_amdgcn_s_wait_tensorcnt)
  __builtin_amdgcn_s_wait_tensorcnt((short)2);
#else
  asm volatile("s_wait_tensorcnt 0x2" ::: "memory");
#endif
}

__device__ __forceinline__ float redmax16(float v) {
  v = fmaxf(v, __shfl_xor(v, 1, 32));
  v = fmaxf(v, __shfl_xor(v, 2, 32));
  v = fmaxf(v, __shfl_xor(v, 4, 32));
  v = fmaxf(v, __shfl_xor(v, 8, 32));
  return v;
}
__device__ __forceinline__ float redsum16(float v) {
  v += __shfl_xor(v, 1, 32);
  v += __shfl_xor(v, 2, 32);
  v += __shfl_xor(v, 4, 32);
  v += __shfl_xor(v, 8, 32);
  return v;
}

// ---------------------------------------------------------------------------
// Weight convert + transpose: in fp32 [K,N] row-major -> out bf16 [N,K]
// ---------------------------------------------------------------------------
__global__ __launch_bounds__(256)
void cvt_transpose_kernel(const float* __restrict__ in, __bf16* __restrict__ out,
                          int K, int N) {
  int i = blockIdx.x * 256 + threadIdx.x;   // linear over N*K (out order)
  int n = i / K;
  int k = i - n * K;
  out[i] = (__bf16)in[(size_t)k * N + n];
}

// ---------------------------------------------------------------------------
// LayerNorm over C=1024, one block (256 thr) per row, bf16 output
// ---------------------------------------------------------------------------
__global__ __launch_bounds__(256)
void ln_kernel(const float* __restrict__ x, const float* __restrict__ g,
               const float* __restrict__ bta, __bf16* __restrict__ out) {
  __shared__ float s1[256];
  __shared__ float s2[256];
  int row = blockIdx.x, tid = threadIdx.x;
  const float* xr = x + (size_t)row * 1024;
  float v[4], s = 0.f, ss = 0.f;
#pragma unroll
  for (int i = 0; i < 4; ++i) {
    v[i] = xr[tid + 256 * i];
    s += v[i]; ss += v[i] * v[i];
  }
  s1[tid] = s; s2[tid] = ss;
  __syncthreads();
  for (int off = 128; off > 0; off >>= 1) {
    if (tid < off) { s1[tid] += s1[tid + off]; s2[tid] += s2[tid + off]; }
    __syncthreads();
  }
  float mu  = s1[0] * (1.f / 1024.f);
  float var = s2[0] * (1.f / 1024.f) - mu * mu;
  float rs  = rsqrtf(var + 1e-5f);
  __bf16* orow = out + (size_t)row * 1024;
#pragma unroll
  for (int i = 0; i < 4; ++i) {
    int c = tid + 256 * i;
    orow[c] = (__bf16)((v[i] - mu) * rs * g[c] + bta[c]);
  }
}

// ---------------------------------------------------------------------------
// bf16 WMMA GEMM:  out[M,N] = A[M,K] @ Bt[N,K]^T (+bias, epilogue)
//   EPI 0: +bias            -> bf16 out
//   EPI 1: +bias, exact GELU-> bf16 out
//   EPI 2: +bias, +residual -> f32 out
// 128x128 tile per 256-thread block; 8 waves in 4(M) x 2(N) grid,
// each wave owns a 32x64 patch = 2x4 WMMA accumulators.
// Dynamic LDS: As[2] @ 0/8K, Bs[2] @ 16K/24K (32 KB), double-buffered.
// Tiles staged by the Tensor Data Mover: wave 0 issues one tensor_load_to_lds
// per 128x32 tile; s_wait_tensorcnt gates the double buffer (TDM ops from one
// wave complete in order), __syncthreads publishes to the other waves.
// ---------------------------------------------------------------------------
template <int EPI>
__global__ __launch_bounds__(256)
void gemm_bf16_kernel(const __bf16* __restrict__ A, const __bf16* __restrict__ Bt,
                      const float* __restrict__ bias, const float* __restrict__ res,
                      float* __restrict__ outF, __bf16* __restrict__ outB,
                      int M, int N, int K) {
  extern __shared__ char smem[];      // 32 KB: As0|As1|Bs0|Bs1

  const int tid = threadIdx.x;
  const int lane = tid & 31;
  const int wave = tid >> 5;
  const int wm = wave >> 1;           // 0..3
  const int wn = wave & 1;            // 0..1
  const int half16 = lane >> 4;       // 0: lanes 0-15, 1: lanes 16-31
  const int mlane = lane & 15;

  const int m0 = blockIdx.y * 128;
  const int n0 = blockIdx.x * 128;

  const char* Ab = (const char*)A;
  const char* Bb = (const char*)Bt;

  v8f acc[2][4] = {};

  if (wave == 0) {                    // prologue: stage K-tile 0 into buffer 0
    tdm_load_2d(0,     Ab + (size_t)m0 * K * 2, 32, 128, (unsigned)K);
    tdm_load_2d(16384, Bb + (size_t)n0 * K * 2, 32, 128, (unsigned)K);
  }

  int buf = 0;
  for (int k0 = 0; k0 < K; k0 += 32, buf ^= 1) {
    const bool more = (k0 + 32 < K);
    if (wave == 0) {
      if (more) {                     // stage next tile into other buffer
        const size_t kB = (size_t)(k0 + 32) * 2;
        tdm_load_2d((unsigned)((buf ^ 1) * 8192),
                    Ab + (size_t)m0 * K * 2 + kB, 32, 128, (unsigned)K);
        tdm_load_2d((unsigned)(16384 + (buf ^ 1) * 8192),
                    Bb + (size_t)n0 * K * 2 + kB, 32, 128, (unsigned)K);
        wait_tensor_2();              // current pair landed; next still in flight
      } else {
        wait_tensor_0();
      }
    }
    __syncthreads();                  // publish staged tile to all waves

    const __bf16* As = (const __bf16*)(smem + buf * 8192);
    const __bf16* Bs = (const __bf16*)(smem + 16384 + buf * 8192);

    // A fragments (contiguous k order: lanes0-15 K0-15, 16-31 K16-31, matches B)
    v16bf af[2];
#pragma unroll
    for (int tm = 0; tm < 2; ++tm) {
      int m = wm * 32 + tm * 16 + mlane;
      af[tm] = *(const v16bf*)&As[m * 32 + (half16 ? 16 : 0)];
    }
    // B fragments + 8 WMMAs
#pragma unroll
    for (int tn = 0; tn < 4; ++tn) {
      int n = wn * 64 + tn * 16 + mlane;
      v16bf bf = *(const v16bf*)&Bs[n * 32 + (half16 ? 16 : 0)];
#pragma unroll
      for (int tm = 0; tm < 2; ++tm)
        acc[tm][tn] = wmma_bf16(af[tm], bf, acc[tm][tn]);
    }
    __syncthreads();                  // done reading buf before it is refilled
  }

  // epilogue: C/D layout row = r + 8*half16, col = lane&15
#pragma unroll
  for (int tm = 0; tm < 2; ++tm) {
#pragma unroll
    for (int tn = 0; tn < 4; ++tn) {
#pragma unroll
      for (int r = 0; r < 8; ++r) {
        int row = m0 + wm * 32 + tm * 16 + r + 8 * half16;
        int col = n0 + wn * 64 + tn * 16 + mlane;
        size_t idx = (size_t)row * N + col;
        float v = acc[tm][tn][r] + bias[col];
        if (EPI == 0) {
          outB[idx] = (__bf16)v;
        } else if (EPI == 1) {
          v = 0.5f * v * (1.0f + erff(v * 0.70710678118654752f));
          outB[idx] = (__bf16)v;
        } else {
          outF[idx] = v + res[idx];
        }
      }
    }
  }
}

// ---------------------------------------------------------------------------
// Flash attention, causal. qkv: bf16 [B*T, 3C]; y: bf16 [B*T, C].
// Block = (b,h,q-block of 64 rows); 4 waves x 16 query rows.
// Dynamic LDS: Kb[64][64] @ 0 (async global->LDS staged), Vt[64][64] @ 8K
// (transposed via registers), per-wave P @ 16K.
// S = Q K^T and O += P V both via bf16 WMMA.
// ---------------------------------------------------------------------------
__global__ __launch_bounds__(128)
void attn_kernel(const __bf16* __restrict__ qkv, __bf16* __restrict__ y) {
  constexpr int Cc = 1024, C3 = 3072, T = 1024;
  extern __shared__ char smem[];      // 24 KB: Kb | Vt | Pw[4]
  __bf16* Kb = (__bf16*)smem;
  __bf16* Vt = (__bf16*)(smem + 8192);

  const int b = blockIdx.y >> 4;
  const int h = blockIdx.y & 15;
  const int q0 = blockIdx.x * 64;
  const int tid = threadIdx.x;
  const int lane = tid & 31;
  const int wave = tid >> 5;
  const int half16 = lane >> 4;
  const int mlane = lane & 15;

  __bf16* Pw = (__bf16*)(smem + 16384) + wave * 1024;

  const size_t base = (size_t)b * T * C3 + (size_t)h * 64;

  // Q fragments for this wave's 16 rows (kept in registers)
  const int tq = q0 + wave * 16 + mlane;
  v16bf qf[2];
#pragma unroll
  for (int s = 0; s < 2; ++s) {
    const __bf16* qp = qkv + base + (size_t)tq * C3 + s * 32;
    qf[s] = *(const v16bf*)(qp + (half16 ? 16 : 0));
  }

  v8f oacc[4] = {};
  float mrow[8], lrow[8];
#pragma unroll
  for (int r = 0; r < 8; ++r) { mrow[r] = -1e30f; lrow[r] = 0.f; }

  const int nkb = blockIdx.x + 1;     // causal: key blocks 0..q-block
  for (int ib = 0; ib < nkb; ++ib) {
    const int kb = ib * 64;

    // stage K block row-major via async global->LDS: 512 x 16B, 4 per thread
#pragma unroll
    for (int it = 0; it < 4; ++it) {
      int c = tid + it * 128;
      int row = c >> 3, off = c & 7;
      async_b128((unsigned)(row * 128 + off * 16),
                 (const char*)qkv + (base + Cc + (size_t)(kb + row) * C3) * 2 + off * 16);
    }
    // stage V transposed through registers: Vt[d][k] = V[k][d]
    {
      int kr = tid >> 1;
      int dbase = (tid & 1) * 32;
      const __bf16* vp = qkv + base + 2 * Cc + (size_t)(kb + kr) * C3 + dbase;
#pragma unroll
      for (int cc = 0; cc < 4; ++cc) {
        v8bf chunk = *(const v8bf*)(vp + cc * 8);
#pragma unroll
        for (int j = 0; j < 8; ++j) Vt[(dbase + cc * 8 + j) * 64 + kr] = chunk[j];
      }
    }
    wait_async_0();
    __syncthreads();

    // S = Q K^T (scaled)
    float sc[4][8];
#pragma unroll
    for (int tn = 0; tn < 4; ++tn) {
      v8f sacc = {};
#pragma unroll
      for (int s = 0; s < 2; ++s) {
        v16bf kf = *(const v16bf*)&Kb[(tn * 16 + mlane) * 64 + s * 32 + (half16 ? 16 : 0)];
        sacc = wmma_bf16(qf[s], kf, sacc);
      }
#pragma unroll
      for (int r = 0; r < 8; ++r) sc[tn][r] = sacc[r] * 0.125f;  // 1/sqrt(64)
    }

    // causal mask (diagonal region only)
    if (kb + 64 > q0) {
#pragma unroll
      for (int tn = 0; tn < 4; ++tn) {
        int kg = kb + tn * 16 + mlane;
#pragma unroll
        for (int r = 0; r < 8; ++r) {
          int qg = q0 + wave * 16 + r + 8 * half16;
          if (kg > qg) sc[tn][r] = -1e30f;
        }
      }
    }

    // online softmax per row (8 rows per lane-half)
#pragma unroll
    for (int r = 0; r < 8; ++r) {
      float mx = sc[0][r];
      mx = fmaxf(mx, sc[1][r]); mx = fmaxf(mx, sc[2][r]); mx = fmaxf(mx, sc[3][r]);
      mx = redmax16(mx);
      float mnew = fmaxf(mrow[r], mx);
      float alpha = __expf(mrow[r] - mnew);
      float psum = 0.f;
#pragma unroll
      for (int tn = 0; tn < 4; ++tn) {
        float p = __expf(sc[tn][r] - mnew);
        sc[tn][r] = p;
        psum += p;
      }
      psum = redsum16(psum);
      lrow[r] = lrow[r] * alpha + psum;
      mrow[r] = mnew;
#pragma unroll
      for (int td = 0; td < 4; ++td) oacc[td][r] *= alpha;
      // store P row to per-wave LDS (C/D layout -> row-major [q][key])
      int m = r + 8 * half16;
#pragma unroll
      for (int tn = 0; tn < 4; ++tn)
        Pw[m * 64 + tn * 16 + mlane] = (__bf16)sc[tn][r];
    }
    __syncthreads();   // make P visible (and keep block convergent)

    // O += P V
#pragma unroll
    for (int td = 0; td < 4; ++td) {
#pragma unroll
      for (int s = 0; s < 2; ++s) {
        v16bf pf = *(const v16bf*)&Pw[mlane * 64 + s * 32 + (half16 ? 16 : 0)];
        v16bf vf = *(const v16bf*)&Vt[(td * 16 + mlane) * 64 + s * 32 + (half16 ? 16 : 0)];
        oacc[td] = wmma_bf16(pf, vf, oacc[td]);
      }
    }
    __syncthreads();   // before next block overwrites Kb/Vt
  }

  // normalize + write y[b*T+q, h*64+d]
#pragma unroll
  for (int r = 0; r < 8; ++r) {
    float invl = 1.0f / lrow[r];
    int tqo = q0 + wave * 16 + r + 8 * half16;
#pragma unroll
    for (int td = 0; td < 4; ++td) {
      int col = h * 64 + td * 16 + mlane;
      y[(size_t)(b * T + tqo) * Cc + col] = (__bf16)(oacc[td][r] * invl);
    }
  }
}

// ---------------------------------------------------------------------------
// Host launch
// ---------------------------------------------------------------------------
extern "C" void kernel_launch(void* const* d_in, const int* in_sizes, int n_in,
                              void* d_out, int out_size, void* d_ws, size_t ws_size,
                              hipStream_t stream) {
  (void)in_sizes; (void)n_in; (void)out_size; (void)ws_size;
  constexpr int T = 1024, Bb = 8;
  constexpr int M = Bb * T;          // 8192 rows

  const float* x       = (const float*)d_in[0];
  const float* ln1_g   = (const float*)d_in[1];
  const float* ln1_b   = (const float*)d_in[2];
  const float* w_attn  = (const float*)d_in[3];
  const float* b_attn  = (const float*)d_in[4];
  const float* w_aproj = (const float*)d_in[5];
  const float* b_aproj = (const float*)d_in[6];
  const float* ln2_g   = (const float*)d_in[7];
  const float* ln2_b   = (const float*)d_in[8];
  const float* w_fc    = (const float*)d_in[9];
  const float* b_fc    = (const float*)d_in[10];
  const float* w_mproj = (const float*)d_in[11];
  const float* b_mproj = (const float*)d_in[12];
  float* out = (float*)d_out;

  // workspace carve-up (all offsets 256B-aligned by construction)
  char* ws = (char*)d_ws;
  __bf16* wattnT  = (__bf16*)ws;  ws += (size_t)3072 * 1024 * 2;
  __bf16* waprojT = (__bf16*)ws;  ws += (size_t)1024 * 1024 * 2;
  __bf16* wfcT    = (__bf16*)ws;  ws += (size_t)4096 * 1024 * 2;
  __bf16* wmprojT = (__bf16*)ws;  ws += (size_t)1024 * 4096 * 2;
  __bf16* h1      = (__bf16*)ws;  ws += (size_t)M * 1024 * 2;
  __bf16* qkv     = (__bf16*)ws;  ws += (size_t)M * 3072 * 2;
  __bf16* yb      = (__bf16*)ws;  ws += (size_t)M * 1024 * 2;
  float*  x1      = (float*)ws;   ws += (size_t)M * 1024 * 4;
  __bf16* h2      = (__bf16*)ws;  ws += (size_t)M * 1024 * 2;
  __bf16* gact    = (__bf16*)ws;  ws += (size_t)M * 4096 * 2;

  // 1) weight convert+transpose (one-time pre-pass per launch)
  cvt_transpose_kernel<<<3072 * 1024 / 256, 256, 0, stream>>>(w_attn,  wattnT,  1024, 3072);
  cvt_transpose_kernel<<<1024 * 1024 / 256, 256, 0, stream>>>(w_aproj, waprojT, 1024, 1024);
  cvt_transpose_kernel<<<4096 * 1024 / 256, 256, 0, stream>>>(w_fc,    wfcT,    1024, 4096);
  cvt_transpose_kernel<<<1024 * 4096 / 256, 256, 0, stream>>>(w_mproj, wmprojT, 4096, 1024);

  // 2) LN1 -> h1 (bf16)
  ln_kernel<<<M, 256, 0, stream>>>(x, ln1_g, ln1_b, h1);

  // 3) QKV = h1 @ w_attn + b_attn  -> bf16 [M,3072]
  gemm_bf16_kernel<0><<<dim3(3072 / 128, M / 128), 256, 32768, stream>>>(
      h1, wattnT, b_attn, nullptr, nullptr, qkv, M, 3072, 1024);

  // 4) causal flash attention -> yb bf16 [M,1024]
  attn_kernel<<<dim3(T / 64, Bb * 16), 128, 24576, stream>>>(qkv, yb);

  // 5) x1 = x + yb @ w_aproj + b_aproj   (f32 residual stream)
  gemm_bf16_kernel<2><<<dim3(1024 / 128, M / 128), 256, 32768, stream>>>(
      yb, waprojT, b_aproj, x, x1, nullptr, M, 1024, 1024);

  // 6) LN2 -> h2 (bf16)
  ln_kernel<<<M, 256, 0, stream>>>(x1, ln2_g, ln2_b, h2);

  // 7) gact = gelu(h2 @ w_fc + b_fc)  -> bf16 [M,4096]
  gemm_bf16_kernel<1><<<dim3(4096 / 128, M / 128), 256, 32768, stream>>>(
      h2, wfcT, b_fc, nullptr, nullptr, gact, M, 4096, 1024);

  // 8) out = x1 + gact @ w_mproj + b_mproj  (f32)
  gemm_bf16_kernel<2><<<dim3(1024 / 128, M / 128), 256, 32768, stream>>>(
      gact, wmprojT, b_mproj, x1, out, nullptr, M, 1024, 4096);
}
